// ECGFFTGlobalPoolNet_6347961664027
// MI455X (gfx1250) — matrix operand order repaired and checked by reading, not compiled, not run on hardware
//
#include <hip/hip_runtime.h>
#include <hip/hip_bf16.h>

// ---------------------------------------------------------------------------
// Types for WMMA fragments (gfx1250, wave32)
// ---------------------------------------------------------------------------
typedef _Float16 v8h  __attribute__((ext_vector_type(8)));
typedef _Float16 v16h __attribute__((ext_vector_type(16)));
typedef float    v8f  __attribute__((ext_vector_type(8)));

// ---------------------------------------------------------------------------
// Problem constants
// ---------------------------------------------------------------------------
#define BATCH 64
#define TLEN  16384
#define C0 64
#define C1 128
#define C2 256
#define L1 8192          // positions after pool0 (input to conv1)
#define L1P 8196         // padded (+2 each side)
#define L2 4096          // positions after pool1 (input to conv2)
#define L2P 4100
#define L3 2048          // pooled conv2 output length
#define K1 320           // Cin*k for conv1 (64*5)
#define K2 640           // Cin*k for conv2 (128*5)
#define NBINS_MAX 8193   // T/2+1
#define BN_EPS 1e-5f

// ws layout (bytes)
static constexpr size_t OFF_H0   = 0;
static constexpr size_t SZ_H0    = (size_t)BATCH * L1P * C0 * 2;
static constexpr size_t OFF_H1   = OFF_H0 + SZ_H0;
static constexpr size_t SZ_H1    = (size_t)BATCH * L2P * C1 * 2;
static constexpr size_t OFF_W1P  = OFF_H1 + SZ_H1;
static constexpr size_t OFF_W2P  = OFF_W1P + (size_t)C1 * K1 * 2;
static constexpr size_t OFF_B1P  = OFF_W2P + (size_t)C2 * K2 * 2;
static constexpr size_t OFF_B2P  = OFF_B1P + 512;
static constexpr size_t OFF_ACC  = OFF_B2P + 1024;
static constexpr size_t OFF_SPEC = OFF_ACC + (size_t)BATCH * C2 * 4;
static constexpr size_t OFF_SPM  = OFF_SPEC + (size_t)BATCH * NBINS_MAX * 4;

// ---------------------------------------------------------------------------
// init: zero time-pool accumulators + the zero-pad halo of h0/h1 (ws is
// poisoned by the harness, so this must run every call)
// ---------------------------------------------------------------------------
__global__ void __launch_bounds__(256) init_kernel(float* __restrict__ acc,
                                                   _Float16* __restrict__ h0,
                                                   _Float16* __restrict__ h1) {
  int idx = blockIdx.x * 256 + threadIdx.x;
  if (idx < BATCH * C2) acc[idx] = 0.f;
  if (idx < BATCH * 4 * C0) {               // h0 halo: p in {0,1,8194,8195}
    int b = idx >> 8; int r = idx & 255; int ps = r >> 6; int c = r & 63;
    int p = (ps < 2) ? ps : (L1 + ps);
    h0[((size_t)b * L1P + p) * C0 + c] = (_Float16)0.f;
  }
  if (idx < BATCH * 4 * C1) {               // h1 halo: p in {0,1,4098,4099}
    int b = idx >> 9; int r = idx & 511; int ps = r >> 7; int c = r & 127;
    int p = (ps < 2) ? ps : (L2 + ps);
    h1[((size_t)b * L2P + p) * C1 + c] = (_Float16)0.f;
  }
}

// ---------------------------------------------------------------------------
// weight prep: fold BN scale into conv1/conv2 weights (f16, K = dt*Cin + ci
// ordering) and fold conv bias + BN shift into one f32 bias per channel
// ---------------------------------------------------------------------------
__global__ void __launch_bounds__(256) prep_kernel(
    const float* __restrict__ w1, const float* __restrict__ cb1,
    const float* __restrict__ g1, const float* __restrict__ be1,
    const float* __restrict__ m1, const float* __restrict__ v1,
    const float* __restrict__ w2, const float* __restrict__ cb2,
    const float* __restrict__ g2, const float* __restrict__ be2,
    const float* __restrict__ m2, const float* __restrict__ v2,
    _Float16* __restrict__ w1p, float* __restrict__ b1p,
    _Float16* __restrict__ w2p, float* __restrict__ b2p) {
  int idx = blockIdx.x * 256 + threadIdx.x;
  if (idx < C1 * K1) {
    int co = idx / K1, k = idx % K1, dt = k / C0, ci = k % C0;
    float s = g1[co] * rsqrtf(v1[co] + BN_EPS);
    w1p[idx] = (_Float16)(w1[(co * C0 + ci) * 5 + dt] * s);
  } else if (idx < C1 * K1 + C2 * K2) {
    int j = idx - C1 * K1;
    int co = j / K2, k = j % K2, dt = k / C1, ci = k % C1;
    float s = g2[co] * rsqrtf(v2[co] + BN_EPS);
    w2p[j] = (_Float16)(w2[(co * C1 + ci) * 5 + dt] * s);
  } else if (idx < C1 * K1 + C2 * K2 + C1) {
    int co = idx - (C1 * K1 + C2 * K2);
    float s = g1[co] * rsqrtf(v1[co] + BN_EPS);
    b1p[co] = cb1[co] * s + be1[co] - m1[co] * s;
  } else if (idx < C1 * K1 + C2 * K2 + C1 + C2) {
    int co = idx - (C1 * K1 + C2 * K2 + C1);
    float s = g2[co] * rsqrtf(v2[co] + BN_EPS);
    b2p[co] = cb2[co] * s + be2[co] - m2[co] * s;
  }
}

// ---------------------------------------------------------------------------
// block0: conv(1->64,k7,pad3) + BN + ReLU + maxpool2  -> h0 [B][L1P][C0] f16
// Cin=1: pure VALU (no matmul structure). blockDim = (64 ch, 4 tp)
// ---------------------------------------------------------------------------
__global__ void __launch_bounds__(256) conv0_kernel(
    const float* __restrict__ x, const float* __restrict__ w0,
    const float* __restrict__ cb0, const float* __restrict__ g0,
    const float* __restrict__ be0, const float* __restrict__ m0,
    const float* __restrict__ v0, _Float16* __restrict__ h0) {
  const int c  = threadIdx.x;
  const int tp = blockIdx.x * 4 + threadIdx.y;   // pooled pos [0,8192)
  const int b  = blockIdx.y;
  const float scale = g0[c] * rsqrtf(v0[c] + BN_EPS);
  const float shift = be0[c] - m0[c] * scale + cb0[c] * scale;
  float w[7];
#pragma unroll
  for (int i = 0; i < 7; ++i) w[i] = w0[c * 7 + i] * scale;
  const float* xb = x + (size_t)b * TLEN;
  const int t0 = tp * 2;
  float xv[8];
#pragma unroll
  for (int i = 0; i < 8; ++i) {
    int ix = t0 - 3 + i;
    xv[i] = (ix >= 0 && ix < TLEN) ? xb[ix] : 0.f;
  }
  float y0 = shift, y1 = shift;
#pragma unroll
  for (int i = 0; i < 7; ++i) { y0 = fmaf(w[i], xv[i], y0); y1 = fmaf(w[i], xv[i + 1], y1); }
  float r = fmaxf(fmaxf(y0, y1), 0.f);
  h0[((size_t)b * L1P + tp + 2) * C0 + c] = (_Float16)r;
}

// ---------------------------------------------------------------------------
// block1 as implicit GEMM with WMMA:
//   D[Cout=128, 16 positions] += W1p[128,320] x patches[320,16]
// 8 waves = 8 M-tiles cover Cout=128. B slab (20 pos x 64 ch) staged in LDS
// once and shared by all waves. Epilogue: bias + ReLU + maxpool2 (shfl_xor),
// write h1 [B][L2P][C1] f16.
// ---------------------------------------------------------------------------
__global__ void __launch_bounds__(256) conv1_kernel(
    const _Float16* __restrict__ h0, const _Float16* __restrict__ w1p,
    const float* __restrict__ b1p, _Float16* __restrict__ h1) {
  const int n0 = blockIdx.x * 16;     // conv position base [0, 8192)
  const int b  = blockIdx.y;
  const int tid = threadIdx.x;
  const int wave = tid >> 5, lane = tid & 31;
  const int lane16 = lane & 15, hlf = lane >> 4;

  __shared__ __align__(16) _Float16 sB[20 * C0];   // 2560 B
  {
    const uint32_t* s32 = (const uint32_t*)(h0 + ((size_t)b * L1P + n0) * C0);
    uint32_t* d32 = (uint32_t*)sB;
    for (int i = tid; i < 20 * C0 / 2; i += 256) d32[i] = s32[i];
  }
  __syncthreads();

  const int mbase = wave * 16;
  v8f acc = {0.f, 0.f, 0.f, 0.f, 0.f, 0.f, 0.f, 0.f};
#pragma unroll
  for (int kk = 0; kk < K1; kk += 32) {
    // A fragment: row = mbase+lane16; halves take K runs of 16
    const int ka = kk + hlf * 16;
    const _Float16* wa = w1p + (size_t)(mbase + lane16) * K1 + ka;
    v8h alo = *(const v8h*)wa;
    v8h ahi = *(const v8h*)(wa + 8);
    v16h a = __builtin_shufflevector(alo, ahi, 0,1,2,3,4,5,6,7,8,9,10,11,12,13,14,15);
    // B fragment: column = n0+lane16; halves take K runs of 16
    const int kb = kk + hlf * 16;
    const int dt = kb >> 6, ci = kb & 63;
    const _Float16* bp = sB + (lane16 + dt) * C0 + ci;
    v8h blo = *(const v8h*)bp;
    v8h bhi = *(const v8h*)(bp + 8);
    v16h bf = __builtin_shufflevector(blo, bhi, 0,1,2,3,4,5,6,7,8,9,10,11,12,13,14,15);
    acc = __builtin_amdgcn_wmma_f32_16x16x32_f16(false, a, false, bf, (short)0, acc,
                                                 false, false);
  }

  // epilogue: bias + relu + pool2 (adjacent columns) + f16 store
  float vals[8];
#pragma unroll
  for (int r = 0; r < 8; ++r) {
    int m = mbase + hlf * 8 + r;
    vals[r] = fmaxf(acc[r] + b1p[m], 0.f);
  }
#pragma unroll
  for (int r = 0; r < 8; ++r)
    vals[r] = fmaxf(vals[r], __shfl_xor(vals[r], 1, 32));
  if ((lane & 1) == 0) {
    int tp = (n0 >> 1) + (lane16 >> 1);
    v8h ov;
#pragma unroll
    for (int r = 0; r < 8; ++r) ov[r] = (_Float16)vals[r];
    *(v8h*)(h1 + ((size_t)b * L2P + tp + 2) * C1 + mbase + hlf * 8) = ov;
  }
}

// ---------------------------------------------------------------------------
// block2: implicit GEMM (Cout=256 -> 2 M-tiles per wave) + bias + ReLU +
// maxpool2 + fused masked global-average pool (xor-tree reduce + one
// global atomicAdd per (b,channel) per block).
// ---------------------------------------------------------------------------
__global__ void __launch_bounds__(256) conv2_kernel(
    const _Float16* __restrict__ h1, const _Float16* __restrict__ w2p,
    const float* __restrict__ b2p, const int* __restrict__ lengths,
    float* __restrict__ acc_out) {
  const int n0 = blockIdx.x * 16;     // conv position base [0, 4096)
  const int b  = blockIdx.y;
  const int tid = threadIdx.x;
  const int wave = tid >> 5, lane = tid & 31;
  const int lane16 = lane & 15, hlf = lane >> 4;

  __shared__ __align__(16) _Float16 sB[20 * C1];   // 5120 B
  {
    const uint32_t* s32 = (const uint32_t*)(h1 + ((size_t)b * L2P + n0) * C1);
    uint32_t* d32 = (uint32_t*)sB;
    for (int i = tid; i < 20 * C1 / 2; i += 256) d32[i] = s32[i];
  }
  __syncthreads();

  const int plen = max(lengths[b] >> 3, 1);   // pooled valid length
  const int m0 = wave * 32;                   // two M-tiles: m0, m0+16
  v8f acc0 = {0.f, 0.f, 0.f, 0.f, 0.f, 0.f, 0.f, 0.f};
  v8f acc1 = {0.f, 0.f, 0.f, 0.f, 0.f, 0.f, 0.f, 0.f};
#pragma unroll 4
  for (int kk = 0; kk < K2; kk += 32) {
    const int kb = kk + hlf * 16;
    const int dt = kb >> 7, ci = kb & 127;
    const _Float16* bp = sB + (lane16 + dt) * C1 + ci;
    v8h blo = *(const v8h*)bp;
    v8h bhi = *(const v8h*)(bp + 8);
    v16h bf = __builtin_shufflevector(blo, bhi, 0,1,2,3,4,5,6,7,8,9,10,11,12,13,14,15);

    const int ka = kk + hlf * 16;
    const _Float16* wa0 = w2p + (size_t)(m0 + lane16) * K2 + ka;
    v8h a0lo = *(const v8h*)wa0;
    v8h a0hi = *(const v8h*)(wa0 + 8);
    v16h a0 = __builtin_shufflevector(a0lo, a0hi, 0,1,2,3,4,5,6,7,8,9,10,11,12,13,14,15);
    const _Float16* wa1 = w2p + (size_t)(m0 + 16 + lane16) * K2 + ka;
    v8h a1lo = *(const v8h*)wa1;
    v8h a1hi = *(const v8h*)(wa1 + 8);
    v16h a1 = __builtin_shufflevector(a1lo, a1hi, 0,1,2,3,4,5,6,7,8,9,10,11,12,13,14,15);

    acc0 = __builtin_amdgcn_wmma_f32_16x16x32_f16(false, a0, false, bf, (short)0, acc0,
                                                  false, false);
    acc1 = __builtin_amdgcn_wmma_f32_16x16x32_f16(false, a1, false, bf, (short)0, acc1,
                                                  false, false);
  }

  // epilogue: bias+relu, pool pairs, mask by tp<plen, xor-tree sum over the
  // 8 pooled positions of this tile, then atomicAdd into acc_out[b][m].
  const int tp = (n0 >> 1) + (lane16 >> 1);
  const float mask = (tp < plen) ? 1.f : 0.f;
#pragma unroll
  for (int tile = 0; tile < 2; ++tile) {
    const int mb = m0 + tile * 16;
#pragma unroll
    for (int r = 0; r < 8; ++r) {
      int m = mb + hlf * 8 + r;
      float v = (tile == 0) ? acc0[r] : acc1[r];
      v = fmaxf(v + b2p[m], 0.f);
      v = fmaxf(v, __shfl_xor(v, 1, 32));   // pooled (duplicated on pair)
      v *= mask;
      v += __shfl_xor(v, 2, 32);
      v += __shfl_xor(v, 4, 32);
      v += __shfl_xor(v, 8, 32);            // sum over 8 distinct pooled pos
      if (lane16 == 0) atomicAdd(&acc_out[b * C2 + m], v);
    }
  }
}

// ---------------------------------------------------------------------------
// frequency branch: per-bin Goertzel DFT (handles arbitrary lengths),
// x streamed through LDS; spec[b][k] = log1p(|X_k|)
// ---------------------------------------------------------------------------
__global__ void __launch_bounds__(256) dft_kernel(
    const float* __restrict__ x, const int* __restrict__ lengths,
    float* __restrict__ spec) {
  const int b = blockIdx.y;
  const int k = blockIdx.x * 256 + threadIdx.x;
  const int L = lengths[b];
  const int nb = (L >> 1) + 1;
  __shared__ float sx[2048];
  const double w = 6.283185307179586 * (double)k / (double)L;
  const float cw = (float)(2.0 * cos(w));
  float s1 = 0.f, s2 = 0.f;
  const float* xb = x + (size_t)b * TLEN;
  for (int base = 0; base < L; base += 2048) {
    const int cnt = min(2048, L - base);
    __syncthreads();
    for (int i = threadIdx.x; i < cnt; i += 256) sx[i] = xb[base + i];
    __syncthreads();
    for (int n = 0; n < cnt; ++n) {
      float s0 = fmaf(cw, s1, sx[n] - s2);
      s2 = s1; s1 = s0;
    }
  }
  if (k < nb) {
    float power = s1 * s1 + s2 * s2 - cw * s1 * s2;
    spec[(size_t)b * NBINS_MAX + k] = log1pf(sqrtf(fmaxf(power, 0.f)));
  }
}

// ---------------------------------------------------------------------------
// exact PyTorch adaptive_avg_pool1d to 256 bins
// ---------------------------------------------------------------------------
__global__ void __launch_bounds__(256) apool_kernel(
    const float* __restrict__ spec, const int* __restrict__ lengths,
    float* __restrict__ spectrum) {
  const int b = blockIdx.x, j = threadIdx.x;
  const int n = (lengths[b] >> 1) + 1;
  const int s = (j * n) >> 8;
  const int e = ((j + 1) * n + 255) >> 8;
  float sum = 0.f;
  for (int i = s; i < e; ++i) sum += spec[(size_t)b * NBINS_MAX + i];
  spectrum[b * 256 + j] = sum / (float)(e - s);
}

// ---------------------------------------------------------------------------
// head: time_vec = acc/plen; freq_vec = relu(spectrum @ Wf^T + bf);
// fused=[time,freq]; hid=relu(fused @ W0^T + b0); out = hid @ W1^T + b1
// One block per sample, 128 threads.
// ---------------------------------------------------------------------------
__global__ void __launch_bounds__(128) head_kernel(
    const float* __restrict__ spectrum, const float* __restrict__ acc,
    const int* __restrict__ lengths, const float* __restrict__ fw,
    const float* __restrict__ fb, const float* __restrict__ hw0,
    const float* __restrict__ hb0, const float* __restrict__ hw1,
    const float* __restrict__ hb1, float* __restrict__ out) {
  const int b = blockIdx.x, tid = threadIdx.x;
  __shared__ float fused[384];
  __shared__ float sp[256];
  __shared__ float hid[128];
  const float denom = (float)max(lengths[b] >> 3, 1);
  for (int c = tid; c < 256; c += 128) fused[c] = acc[b * C2 + c] / denom;
  for (int j = tid; j < 256; j += 128) sp[j] = spectrum[b * 256 + j];
  __syncthreads();
  {
    float s = fb[tid];
    for (int j = 0; j < 256; ++j) s = fmaf(sp[j], fw[tid * 256 + j], s);
    fused[256 + tid] = fmaxf(s, 0.f);
  }
  __syncthreads();
  {
    float s = hb0[tid];
    for (int d = 0; d < 384; ++d) s = fmaf(fused[d], hw0[tid * 384 + d], s);
    hid[tid] = fmaxf(s, 0.f);
  }
  __syncthreads();
  if (tid < 4) {
    float s = hb1[tid];
    for (int h = 0; h < 128; ++h) s = fmaf(hid[h], hw1[tid * 128 + h], s);
    out[b * 4 + tid] = s;
  }
}

// ---------------------------------------------------------------------------
extern "C" void kernel_launch(void* const* d_in, const int* in_sizes, int n_in,
                              void* d_out, int out_size, void* d_ws, size_t ws_size,
                              hipStream_t stream) {
  const float* x       = (const float*)d_in[0];
  const int*   lengths = (const int*)d_in[1];
  const float* w0  = (const float*)d_in[2];
  const float* cb0 = (const float*)d_in[3];
  const float* g0  = (const float*)d_in[4];
  const float* be0 = (const float*)d_in[5];
  const float* m0  = (const float*)d_in[6];
  const float* v0  = (const float*)d_in[7];
  const float* w1  = (const float*)d_in[8];
  const float* cb1 = (const float*)d_in[9];
  const float* g1  = (const float*)d_in[10];
  const float* be1 = (const float*)d_in[11];
  const float* m1  = (const float*)d_in[12];
  const float* v1  = (const float*)d_in[13];
  const float* w2  = (const float*)d_in[14];
  const float* cb2 = (const float*)d_in[15];
  const float* g2  = (const float*)d_in[16];
  const float* be2 = (const float*)d_in[17];
  const float* m2  = (const float*)d_in[18];
  const float* v2  = (const float*)d_in[19];
  const float* fw  = (const float*)d_in[20];
  const float* fb  = (const float*)d_in[21];
  const float* hw0 = (const float*)d_in[22];
  const float* hb0 = (const float*)d_in[23];
  const float* hw1 = (const float*)d_in[24];
  const float* hb1 = (const float*)d_in[25];
  float* out = (float*)d_out;

  char* ws = (char*)d_ws;
  _Float16* h0  = (_Float16*)(ws + OFF_H0);
  _Float16* h1  = (_Float16*)(ws + OFF_H1);
  _Float16* w1p = (_Float16*)(ws + OFF_W1P);
  _Float16* w2p = (_Float16*)(ws + OFF_W2P);
  float* b1p = (float*)(ws + OFF_B1P);
  float* b2p = (float*)(ws + OFF_B2P);
  float* acc = (float*)(ws + OFF_ACC);
  float* spec = (float*)(ws + OFF_SPEC);
  float* spectrum = (float*)(ws + OFF_SPM);

  init_kernel<<<128, 256, 0, stream>>>(acc, h0, h1);
  prep_kernel<<<(C1 * K1 + C2 * K2 + C1 + C2 + 255) / 256, 256, 0, stream>>>(
      w1, cb1, g1, be1, m1, v1, w2, cb2, g2, be2, m2, v2, w1p, b1p, w2p, b2p);
  conv0_kernel<<<dim3(L1 / 4, BATCH), dim3(64, 4), 0, stream>>>(
      x, w0, cb0, g0, be0, m0, v0, h0);
  conv1_kernel<<<dim3(L1 / 16, BATCH), 256, 0, stream>>>(h0, w1p, b1p, h1);
  conv2_kernel<<<dim3(L2 / 16, BATCH), 256, 0, stream>>>(h1, w2p, b2p, lengths, acc);
  dft_kernel<<<dim3((NBINS_MAX + 255) / 256, BATCH), 256, 0, stream>>>(x, lengths, spec);
  apool_kernel<<<BATCH, 256, 0, stream>>>(spec, lengths, spectrum);
  head_kernel<<<BATCH, 128, 0, stream>>>(spectrum, acc, lengths, fw, fb, hw0, hb0,
                                         hw1, hb1, out);
}
